// Encoder_29197187678745
// MI455X (gfx1250) — compile-verified, matrix operand order checked by reading, not run on hardware
//
#include <hip/hip_runtime.h>
#include <hip/hip_bf16.h>

typedef unsigned short u16;
typedef __attribute__((ext_vector_type(16))) __bf16 v16bf;
typedef __attribute__((ext_vector_type(8)))  float  v8f;

#define DDIM 1024
#define SDIM 2048
#define BDIM 2
#define HDIM 16
#define DH   64
#define FFD  4096
#define MROWS 4096   // B*S

// ---------- helpers ----------
__device__ __forceinline__ u16 f2bf(float f) {
    unsigned u = __float_as_uint(f);
    unsigned r = ((u >> 16) & 1u) + 0x7FFFu;
    return (u16)((u + r) >> 16);
}

union Frag16 { uint4 q[2]; v16bf v; };

// Load a 16x32 bf16 fragment (A or B operand) from a row-major tile.
// Lane mapping per CDNA5 ISA: lane<16 -> row=lane, k in {0..7,16..23};
// lane>=16 -> row=lane-16, k in {8..15,24..31}.
__device__ __forceinline__ v16bf ld_frag(const u16* t, int row0, int stride,
                                         int coloff, int lane) {
    int r = lane & 15, hi = lane >> 4;
    const u16* p = t + (size_t)(row0 + r) * stride + coloff + hi * 8;
    Frag16 f;
    f.q[0] = *(const uint4*)(p);
    f.q[1] = *(const uint4*)(p + 16);
    return f.v;
}

__device__ __forceinline__ v8f wmma_bf16(v16bf a, v16bf b, v8f c) {
    return __builtin_amdgcn_wmma_f32_16x16x32_bf16(
        false, a, false, b, (short)0, c, false, false);
}

// CDNA5 async memory->LDS copy (16B per lane), tracked by ASYNCcnt.
// GV mode: vdst = LDS byte address (low 32 bits of generic shared ptr per
// the LDS aperture rule), vaddr = 64-bit global address.
__device__ __forceinline__ void async_ld_b128(u16* lds_ptr, const u16* gptr) {
    unsigned lds_off = (unsigned)(unsigned long long)lds_ptr;
    asm volatile("global_load_async_to_lds_b128 %0, %1, off"
                 :: "v"(lds_off), "v"(gptr) : "memory");
}

// ---------- fp32 -> bf16 conversion ----------
__global__ __launch_bounds__(256) void cvt_bf16_kernel(
    const float* __restrict__ in, u16* __restrict__ out, int n4) {
    int i = blockIdx.x * 256 + threadIdx.x;
    if (i >= n4) return;
    float4 v = *(const float4*)(in + (size_t)i * 4);
    u16 o[4] = { f2bf(v.x), f2bf(v.y), f2bf(v.z), f2bf(v.w) };
    *(uint2*)(out + (size_t)i * 4) = *(const uint2*)o;
}

// ---------- bf16 WMMA GEMM: C[M,N] = act(A[M,K] @ W[N,K]^T + bias) (+res) ----------
// Block: 256 threads (8 waves), 128x128 tile, K-step 32, double-buffered LDS
// filled by GLOBAL_LOAD_ASYNC_TO_LDS_B128 (ASYNCcnt-pipelined prefetch).
// Waves: 4 (m) x 2 (n); each wave computes 32x64 = 2x4 fragments.
template <bool BIAS, bool RELU, bool RESID, bool OUTBF>
__global__ __launch_bounds__(256) void gemm_bf16_kernel(
    const u16* __restrict__ A, const u16* __restrict__ W,
    const float* __restrict__ bias, const float* __restrict__ resid,
    float* __restrict__ outF, u16* __restrict__ outB,
    int Mdim, int Ndim, int Kdim) {
    __shared__ u16 As[2][128 * 40];
    __shared__ u16 Bs[2][128 * 40];

    const int tid  = threadIdx.x;
    const int lane = tid & 31;
    const int wave = tid >> 5;
    const int wm   = wave >> 1;   // 0..3
    const int wn   = wave & 1;    // 0..1
    const int bm   = blockIdx.y * 128;
    const int bn   = blockIdx.x * 128;
    const int hi   = lane >> 4;

    v8f acc[2][4] = {};

    // per-thread staging coordinates (2 x b128 per matrix per K-step)
    const int row0 = tid >> 2,         cg0 = tid & 3;           // idx = tid
    const int row1 = (tid + 256) >> 2, cg1 = (tid + 256) & 3;   // idx = tid+256

    auto stage = [&](int buf, int k0) {
        async_ld_b128(&As[buf][row0 * 40 + cg0 * 8],
                      A + (size_t)(bm + row0) * Kdim + k0 + cg0 * 8);
        async_ld_b128(&As[buf][row1 * 40 + cg1 * 8],
                      A + (size_t)(bm + row1) * Kdim + k0 + cg1 * 8);
        async_ld_b128(&Bs[buf][row0 * 40 + cg0 * 8],
                      W + (size_t)(bn + row0) * Kdim + k0 + cg0 * 8);
        async_ld_b128(&Bs[buf][row1 * 40 + cg1 * 8],
                      W + (size_t)(bn + row1) * Kdim + k0 + cg1 * 8);
    };

    const int KT = Kdim / 32;
    stage(0, 0);

    for (int kt = 0; kt < KT; ++kt) {
        const int buf = kt & 1;
        if (kt + 1 < KT) {
            stage(buf ^ 1, (kt + 1) * 32);
            // 4 newest (next tile) may stay in flight; current tile complete.
            asm volatile("s_wait_asynccnt 0x4" ::: "memory");
        } else {
            asm volatile("s_wait_asynccnt 0x0" ::: "memory");
        }
        __syncthreads();

        v16bf af[2], bf[4];
#pragma unroll
        for (int i = 0; i < 2; ++i)
            af[i] = ld_frag(&As[buf][0], wm * 32 + i * 16, 40, 0, lane);
#pragma unroll
        for (int j = 0; j < 4; ++j)
            bf[j] = ld_frag(&Bs[buf][0], wn * 64 + j * 16, 40, 0, lane);
#pragma unroll
        for (int i = 0; i < 2; ++i)
#pragma unroll
            for (int j = 0; j < 4; ++j)
                acc[i][j] = wmma_bf16(af[i], bf[j], acc[i][j]);

        __syncthreads();   // all waves done reading buf before it is re-filled
    }

    // Epilogue. C layout: row = rr + 8*hi, col = lane&15.
#pragma unroll
    for (int i = 0; i < 2; ++i)
#pragma unroll
        for (int j = 0; j < 4; ++j)
#pragma unroll
            for (int rr = 0; rr < 8; ++rr) {
                int row = bm + wm * 32 + i * 16 + rr + 8 * hi;
                int col = bn + wn * 64 + j * 16 + (lane & 15);
                float v = acc[i][j][rr];
                if (BIAS)  v += bias[col];
                if (RELU)  v = fmaxf(v, 0.0f);
                if (RESID) v += resid[(size_t)row * Ndim + col];
                if (OUTBF) outB[(size_t)row * Ndim + col] = f2bf(v);
                else       outF[(size_t)row * Ndim + col] = v;
            }
}

// ---------- flash attention with relu(scores) then softmax ----------
// grid: (S/128, H, B). Block 256 thr = 8 waves; each wave owns 16 queries.
__global__ __launch_bounds__(256) void attn_kernel(
    const u16* __restrict__ q, const u16* __restrict__ k,
    const u16* __restrict__ v, u16* __restrict__ ctx) {
    __shared__ u16 Kt[32 * 72];        // [key][feat]    (async-filled)
    __shared__ u16 Vt[64 * 40];        // [feat][key]    (manual transpose)
    __shared__ u16 Pt[8][16 * 40];     // per-wave P buffer [row][key]

    const int tid  = threadIdx.x;
    const int lane = tid & 31;
    const int wave = tid >> 5;
    const int hi   = lane >> 4;
    const int h    = blockIdx.y;
    const int b    = blockIdx.z;
    const int qrow0 = b * SDIM + blockIdx.x * 128 + wave * 16;

    // Q fragments (16 x 64 per wave) straight from global bf16
    v16bf qa[2];
#pragma unroll
    for (int d = 0; d < 2; ++d)
        qa[d] = ld_frag(q + (size_t)qrow0 * DDIM + h * DH + d * 32, 0, DDIM, 0, lane);

    v8f o[4] = {};
    float Mx[8], Lr[8];
#pragma unroll
    for (int rr = 0; rr < 8; ++rr) { Mx[rr] = -1e30f; Lr[rr] = 0.0f; }

    const int srow = tid >> 3, scg = tid & 7;   // 32 rows x 8 groups of 8

    for (int kb = 0; kb < SDIM / 32; ++kb) {
        __syncthreads();
        const int krow = b * SDIM + kb * 32 + srow;
        // K tile: direct async memory->LDS (one b128 per thread)
        async_ld_b128(&Kt[srow * 72 + scg * 8],
                      k + (size_t)krow * DDIM + h * DH + scg * 8);
        // V tile: load + transpose-scatter into LDS
        {
            uint4 vvv = *(const uint4*)(v + (size_t)krow * DDIM + h * DH + scg * 8);
            const u16* vs = (const u16*)&vvv;
#pragma unroll
            for (int j = 0; j < 8; ++j)
                Vt[(scg * 8 + j) * 40 + srow] = vs[j];
        }
        asm volatile("s_wait_asynccnt 0x0" ::: "memory");
        __syncthreads();

        // scores: 16 queries x 32 keys
        v8f s[2] = {};
#pragma unroll
        for (int kc = 0; kc < 2; ++kc)
#pragma unroll
            for (int d = 0; d < 2; ++d)
                s[kc] = wmma_bf16(qa[d], ld_frag(Kt, kc * 16, 72, d * 32, lane), s[kc]);

        // scale + relu + streaming softmax
#pragma unroll
        for (int rr = 0; rr < 8; ++rr) {
            float s0 = fmaxf(s[0][rr] * 0.125f, 0.0f);
            float s1 = fmaxf(s[1][rr] * 0.125f, 0.0f);
            float mloc = fmaxf(s0, s1);
#pragma unroll
            for (int off = 8; off; off >>= 1)
                mloc = fmaxf(mloc, __shfl_xor(mloc, off, 32));
            float mnew  = fmaxf(Mx[rr], mloc);
            float alpha = __expf(Mx[rr] - mnew);
            float p0 = __expf(s0 - mnew);
            float p1 = __expf(s1 - mnew);
            float sum = p0 + p1;
#pragma unroll
            for (int off = 8; off; off >>= 1)
                sum += __shfl_xor(sum, off, 32);
            Lr[rr] = Lr[rr] * alpha + sum;
            Mx[rr] = mnew;
            s[0][rr] = p0;
            s[1][rr] = p1;
#pragma unroll
            for (int c = 0; c < 4; ++c)
                o[c][rr] *= alpha;
        }

        // P (C-layout) -> wave-private LDS -> reload in A-fragment layout
        u16* pb = Pt[wave];
#pragma unroll
        for (int kc = 0; kc < 2; ++kc)
#pragma unroll
            for (int rr = 0; rr < 8; ++rr)
                pb[(rr + 8 * hi) * 40 + kc * 16 + (lane & 15)] = f2bf(s[kc][rr]);

        v16bf pa = ld_frag(pb, 0, 40, 0, lane);
#pragma unroll
        for (int c = 0; c < 4; ++c)
            o[c] = wmma_bf16(pa, ld_frag(Vt, c * 16, 40, 0, lane), o[c]);
    }

    // normalize and emit ctx as bf16
#pragma unroll
    for (int c = 0; c < 4; ++c)
#pragma unroll
        for (int rr = 0; rr < 8; ++rr) {
            float val = o[c][rr] / Lr[rr];
            int row = qrow0 + rr + 8 * hi;
            int col = h * DH + c * 16 + (lane & 15);
            ctx[(size_t)row * DDIM + col] = f2bf(val);
        }
}

// ---------- LayerNorm over D=1024 (input already holds x + residual) ----------
__global__ __launch_bounds__(256) void ln_kernel(
    const float* __restrict__ x, const float* __restrict__ g,
    const float* __restrict__ bta, float* __restrict__ yf,
    u16* __restrict__ yb) {
    __shared__ float sh[16];
    const int row = blockIdx.x, tid = threadIdx.x;
    const float* xr = x + (size_t)row * DDIM;
    float4 xv = *(const float4*)(xr + tid * 4);
    float s = xv.x + xv.y + xv.z + xv.w;
    float qq = xv.x * xv.x + xv.y * xv.y + xv.z * xv.z + xv.w * xv.w;
#pragma unroll
    for (int off = 16; off; off >>= 1) {
        s  += __shfl_xor(s, off, 32);
        qq += __shfl_xor(qq, off, 32);
    }
    int wv = tid >> 5, ln = tid & 31;
    if (ln == 0) { sh[wv] = s; sh[8 + wv] = qq; }
    __syncthreads();
    if (tid == 0) {
        float a = 0.f, b2 = 0.f;
        for (int i = 0; i < 8; ++i) { a += sh[i]; b2 += sh[8 + i]; }
        sh[0] = a; sh[8] = b2;
    }
    __syncthreads();
    float mu  = sh[0] * (1.0f / DDIM);
    float var = sh[8] * (1.0f / DDIM) - mu * mu;
    float inv = rsqrtf(var + 1e-5f);
    const float* xp = (const float*)&xv;
#pragma unroll
    for (int j = 0; j < 4; ++j) {
        int col = tid * 4 + j;
        float yv = (xp[j] - mu) * inv * g[col] + bta[col];
        yf[(size_t)row * DDIM + col] = yv;
        if (yb) yb[(size_t)row * DDIM + col] = f2bf(yv);
    }
}

// ---------- host ----------
extern "C" void kernel_launch(void* const* d_in, const int* in_sizes, int n_in,
                              void* d_out, int out_size, void* d_ws, size_t ws_size,
                              hipStream_t stream) {
    const float* X    = (const float*)d_in[0];
    const float* Wq   = (const float*)d_in[1];
    const float* Wk   = (const float*)d_in[2];
    const float* Wo   = (const float*)d_in[3];
    const float* ln1g = (const float*)d_in[4];
    const float* ln1b = (const float*)d_in[5];
    const float* ln2g = (const float*)d_in[6];
    const float* ln2b = (const float*)d_in[7];
    const float* W1   = (const float*)d_in[8];
    const float* b1   = (const float*)d_in[9];
    const float* W2   = (const float*)d_in[10];
    const float* b2   = (const float*)d_in[11];

    char* ws = (char*)d_ws;
    size_t off = 0;
    auto alloc = [&](size_t bytes) {
        char* p = ws + off;
        off += (bytes + 255) & ~size_t(255);
        return p;
    };
    u16*   Xb   = (u16*)alloc((size_t)MROWS * DDIM * 2);
    u16*   Wqb  = (u16*)alloc((size_t)DDIM * DDIM * 2);
    u16*   Wkb  = (u16*)alloc((size_t)DDIM * DDIM * 2);
    u16*   Wob  = (u16*)alloc((size_t)DDIM * DDIM * 2);
    u16*   W1b  = (u16*)alloc((size_t)FFD * DDIM * 2);
    u16*   W2b  = (u16*)alloc((size_t)DDIM * FFD * 2);
    u16*   qb   = (u16*)alloc((size_t)MROWS * DDIM * 2);
    u16*   kb   = (u16*)alloc((size_t)MROWS * DDIM * 2);
    u16*   vb   = (u16*)alloc((size_t)MROWS * DDIM * 2);
    u16*   ctxb = (u16*)alloc((size_t)MROWS * DDIM * 2);
    float* ares = (float*)alloc((size_t)MROWS * DDIM * 4);
    float* yf   = (float*)alloc((size_t)MROWS * DDIM * 4);
    u16*   yb   = (u16*)alloc((size_t)MROWS * DDIM * 2);
    u16*   hb   = (u16*)alloc((size_t)MROWS * FFD * 2);
    float* fres = (float*)alloc((size_t)MROWS * DDIM * 4);

    auto cvt = [&](const float* in, u16* out, size_t n) {
        int n4 = (int)(n / 4);
        cvt_bf16_kernel<<<(n4 + 255) / 256, 256, 0, stream>>>(in, out, n4);
    };
    cvt(X,  Xb,  (size_t)MROWS * DDIM);
    cvt(Wq, Wqb, (size_t)DDIM * DDIM);
    cvt(Wk, Wkb, (size_t)DDIM * DDIM);
    cvt(Wo, Wob, (size_t)DDIM * DDIM);
    cvt(W1, W1b, (size_t)FFD * DDIM);
    cvt(W2, W2b, (size_t)DDIM * FFD);

    // Q, K, V projections (V uses Wo per reference) -> bf16
    dim3 gproj(DDIM / 128, MROWS / 128);
    gemm_bf16_kernel<false, false, false, true><<<gproj, 256, 0, stream>>>(
        Xb, Wqb, nullptr, nullptr, nullptr, qb, MROWS, DDIM, DDIM);
    gemm_bf16_kernel<false, false, false, true><<<gproj, 256, 0, stream>>>(
        Xb, Wkb, nullptr, nullptr, nullptr, kb, MROWS, DDIM, DDIM);
    gemm_bf16_kernel<false, false, false, true><<<gproj, 256, 0, stream>>>(
        Xb, Wob, nullptr, nullptr, nullptr, vb, MROWS, DDIM, DDIM);

    // attention -> ctx (bf16)
    attn_kernel<<<dim3(SDIM / 128, HDIM, BDIM), 256, 0, stream>>>(qb, kb, vb, ctxb);

    // att_out = ctx @ Wo^T + X  (fp32)
    gemm_bf16_kernel<false, false, true, false><<<gproj, 256, 0, stream>>>(
        ctxb, Wob, nullptr, X, ares, nullptr, MROWS, DDIM, DDIM);

    // y = LN1(att_out + X)  -> fp32 + bf16
    ln_kernel<<<MROWS, 256, 0, stream>>>(ares, ln1g, ln1b, yf, yb);

    // h = relu(y @ W1^T + b1) -> bf16
    dim3 gffn1(FFD / 128, MROWS / 128);
    gemm_bf16_kernel<true, true, false, true><<<gffn1, 256, 0, stream>>>(
        yb, W1b, b1, nullptr, nullptr, hb, MROWS, FFD, DDIM);

    // f = h @ W2^T + b2 + y -> fp32
    gemm_bf16_kernel<true, false, true, false><<<gproj, 256, 0, stream>>>(
        hb, W2b, b2, yf, fres, nullptr, MROWS, DDIM, FFD);

    // out = LN2(f + y)
    ln_kernel<<<MROWS, 256, 0, stream>>>(fres, ln2g, ln2b, (float*)d_out, nullptr);
}
